// TMACWeightOnlyQuantizedLinear_60799557042474
// MI455X (gfx1250) — 8-bit matrix operand order repaired and checked by reading, compiled, not run on hardware
//
#include <hip/hip_runtime.h>
#include <stdint.h>


typedef int   v8i __attribute__((ext_vector_type(8)));
typedef int   v4i __attribute__((ext_vector_type(4)));
typedef int   v2i __attribute__((ext_vector_type(2)));
typedef float v4f __attribute__((ext_vector_type(4)));

static constexpr int N    = 8;
static constexpr int M    = 8192;
static constexpr int K    = 8192;
static constexpr int NG   = K / 64;
static constexpr int KM4  = (M / 4) * K;


__global__ void __launch_bounds__(256)
k_abs_partial(const float* __restrict__ w, float* __restrict__ partial) {
    const int tid = threadIdx.x;
    const int gsz = gridDim.x * blockDim.x;
    float s = 0.0f;
    const v4f* w4 = (const v4f*)w;
    for (int i = blockIdx.x * blockDim.x + tid; i < KM4; i += gsz) {
        v4f v = w4[i];
        s += fabsf(v.x) + fabsf(v.y) + fabsf(v.z) + fabsf(v.w);
    }
    __shared__ float red[256];
    red[tid] = s;
    __syncthreads();
    for (int o = 128; o > 0; o >>= 1) {
        if (tid < o) red[tid] += red[tid + o];
        __syncthreads();
    }
    if (tid == 0) partial[blockIdx.x] = red[0];
}

__global__ void __launch_bounds__(256)
k_finalize(const float* __restrict__ partial, float* __restrict__ scales) {
    const int tid = threadIdx.x;
    float s = partial[tid] + partial[tid + 256] + partial[tid + 512] + partial[tid + 768];
    __shared__ float red[256];
    red[tid] = s;
    __syncthreads();
    for (int o = 128; o > 0; o >>= 1) {
        if (tid < o) red[tid] += red[tid + o];
        __syncthreads();
    }
    if (tid == 0) {
        float mean = red[0] / 67108864.0f;
        float wd   = fmaxf(mean, 1e-5f);
        scales[0]  = 1.0f / wd;
        scales[1]  = wd;
    }
}

__global__ void __launch_bounds__(256)
k_act_quant(const float* __restrict__ x, float* __restrict__ aScale,
            int8_t* __restrict__ qx) {
    const int t   = blockIdx.x * 256 + threadIdx.x;
    const int idx = t & 1023;
    const int n   = idx / NG;
    const int g   = idx % NG;
    if (t < 1024) {
        const float* xp = x + (size_t)n * K + g * 64;
        float amax = 0.0f;
        #pragma unroll 8
        for (int i = 0; i < 64; ++i) amax = fmaxf(amax, fabsf(xp[i]));
        float sc = fmaxf(amax * (1.0f / 127.0f), 1e-8f);
        aScale[n * NG + g] = sc;
        float inv = 1.0f / sc;
        uint32_t* qp = (uint32_t*)(qx + (size_t)n * K + g * 64);
        #pragma unroll 4
        for (int i = 0; i < 16; ++i) {
            uint32_t p = 0;
            #pragma unroll
            for (int b = 0; b < 4; ++b) {
                int q = __float2int_rn(xp[i * 4 + b] * inv);
                q = max(-127, min(127, q));
                p |= ((uint32_t)(q & 0xFF)) << (8 * b);
            }
            qp[i] = p;
        }
    } else {
        const int nn = 8 + n;
        aScale[nn * NG + g] = 0.0f;
        v4i z = {0, 0, 0, 0};
        v4i* qp = (v4i*)(qx + (size_t)nn * K + g * 64);
        qp[0] = z; qp[1] = z; qp[2] = z; qp[3] = z;
    }
}

__global__ void __launch_bounds__(256)
k_wquant(const float* __restrict__ w, const float* __restrict__ scales,
         uint32_t* __restrict__ qw) {
    const float sw = scales[0];
    const int gsz = gridDim.x * blockDim.x;
    const v4f* w4 = (const v4f*)w;
    for (int i = blockIdx.x * blockDim.x + threadIdx.x; i < KM4; i += gsz) {
        v4f v = w4[i];
        int q0 = max(-1, min(1, __float2int_rn(v.x * sw)));
        int q1 = max(-1, min(1, __float2int_rn(v.y * sw)));
        int q2 = max(-1, min(1, __float2int_rn(v.z * sw)));
        int q3 = max(-1, min(1, __float2int_rn(v.w * sw)));
        qw[i] = (uint32_t)(q0 & 0xFF) | ((uint32_t)(q1 & 0xFF) << 8) |
                ((uint32_t)(q2 & 0xFF) << 16) | ((uint32_t)(q3 & 0xFF) << 24);
    }
}

__device__ __forceinline__ v8i ldA(const int8_t* ap) {
    v2i a0 = *(const v2i*)(ap);
    v2i a1 = *(const v2i*)(ap + 8);
    v2i a2 = *(const v2i*)(ap + 32);
    v2i a3 = *(const v2i*)(ap + 40);
    v8i A;
    A[0] = a0[0]; A[1] = a0[1]; A[2] = a1[0]; A[3] = a1[1];
    A[4] = a2[0]; A[5] = a2[1]; A[6] = a3[0]; A[7] = a3[1];
    return A;
}
__device__ __forceinline__ v8i ldB(const int8_t* bp) {
    v4i b0 = *(const v4i*)(bp);
    v4i b1 = *(const v4i*)(bp + 32);
    v8i B;
    B[0] = b0[0]; B[1] = b0[1]; B[2] = b0[2]; B[3] = b0[3];
    B[4] = b1[0]; B[5] = b1[1]; B[6] = b1[2]; B[7] = b1[3];
    return B;
}
__device__ __forceinline__ v8i wmma_iu8(v8i A, v8i B) {
    v8i C = {0, 0, 0, 0, 0, 0, 0, 0};
    return __builtin_amdgcn_wmma_i32_16x16x64_iu8(true, A, true, B, C, false, false);
}
__device__ __forceinline__ void consumeD(v8i D, float sc, float* acc) {
    #pragma unroll
    for (int v = 0; v < 8; ++v) acc[v] += sc * (float)D[v];
}

__global__ void __launch_bounds__(128)
k_gemm(const int8_t* __restrict__ qw, const int8_t* __restrict__ qx,
       const float* __restrict__ aScale, const float* __restrict__ scales,
       float* __restrict__ out) {
    const int lane = threadIdx.x & 31;
    const int wave = threadIdx.x >> 5;
    const int tile = blockIdx.x * 4 + wave;
    const int m0   = tile * 16;
    const int r    = lane & 15;
    const int hi   = lane >> 4;

    const int8_t* aBase = qw + (size_t)(m0 + r) * K + hi * 16;
    const int8_t* bBase = qx + (size_t)r * K + hi * 16;
    const float*  sBase = aScale + r * NG;

    float acc0[8], acc1[8];
    #pragma unroll
    for (int v = 0; v < 8; ++v) { acc0[v] = 0.0f; acc1[v] = 0.0f; }

    v8i D0 = wmma_iu8(ldA(aBase), ldB(bBase));
    v8i D1 = wmma_iu8(ldA(aBase + 64), ldB(bBase + 64));

    for (int g = 2; g < NG; g += 2) {
        v8i Ae = ldA(aBase + g * 64);
        v8i Be = ldB(bBase + g * 64);
        consumeD(D0, sBase[g - 2], acc0);
        D0 = wmma_iu8(Ae, Be);

        v8i Ao = ldA(aBase + (g + 1) * 64);
        v8i Bo = ldB(bBase + (g + 1) * 64);
        consumeD(D1, sBase[g - 1], acc1);
        D1 = wmma_iu8(Ao, Bo);
    }
    consumeD(D0, sBase[NG - 2], acc0);
    consumeD(D1, sBase[NG - 1], acc1);

    if (r < N) {
        float wd = scales[1];
        float* op = out + (size_t)r * M + m0 + hi * 8;
        v4f o0 = {(acc0[0] + acc1[0]) * wd, (acc0[1] + acc1[1]) * wd,
                  (acc0[2] + acc1[2]) * wd, (acc0[3] + acc1[3]) * wd};
        v4f o1 = {(acc0[4] + acc1[4]) * wd, (acc0[5] + acc1[5]) * wd,
                  (acc0[6] + acc1[6]) * wd, (acc0[7] + acc1[7]) * wd};
        *(v4f*)op       = o0;
        *((v4f*)op + 1) = o1;
    }
}

extern "C" void kernel_launch(void* const* d_in, const int* in_sizes, int n_in,
                              void* d_out, int out_size, void* d_ws, size_t ws_size,
                              hipStream_t stream) {
    const float* x = (const float*)d_in[0];
    const float* w = (const float*)d_in[1];
    float* out = (float*)d_out;

    char* ws = (char*)d_ws;
    float*    absPartial = (float*)(ws);
    float*    scales     = (float*)(ws + 4096);
    float*    aScale     = (float*)(ws + 8192);
    int8_t*   qx         = (int8_t*)(ws + 16384);
    int8_t*   qw         = (int8_t*)(ws + 147456);

    k_abs_partial<<<1024, 256, 0, stream>>>(w, absPartial);
    k_finalize  <<<1,    256, 0, stream>>>(absPartial, scales);
    k_act_quant <<<8,    256, 0, stream>>>(x, aScale, qx);
    k_wquant    <<<2048, 256, 0, stream>>>(w, scales, (uint32_t*)qw);
    k_gemm      <<<128,  128, 0, stream>>>(qw, qx, aScale, scales, out);
}
